// EinsteinLocalAttention_25039659336141
// MI455X (gfx1250) — compile-verified, matrix operand order checked by reading, not run on hardware
//
#include <hip/hip_runtime.h>

// ---------------------------------------------------------------------------
// EinsteinLocalAttention for MI455X (gfx1250, wave32, WMMA + TDM)
//   B=2, N=4096, C=1024, H=16, D=64, WIN=20 (21 taps)
// ---------------------------------------------------------------------------

typedef __attribute__((ext_vector_type(16))) __bf16 v16bf;
typedef __attribute__((ext_vector_type(8)))  float  v8f;
typedef unsigned int u32x4 __attribute__((ext_vector_type(4)));
typedef int          i32x4 __attribute__((ext_vector_type(4)));
typedef int          i32x8 __attribute__((ext_vector_type(8)));

#define BATCH 2
#define NSEQ  4096
#define CDIM  1024
#define HEADS 16
#define DHEAD 64
#define W2    10
#define WINSZ 21

#define TILE_M 128
#define TILE_N 64
#define KSTEP  32

#if defined(__has_builtin)
#  if __has_builtin(__builtin_amdgcn_tensor_load_to_lds)
#    define HAVE_TDM 1
#  else
#    define HAVE_TDM 0
#  endif
#else
#  define HAVE_TDM 0
#endif

#if defined(__has_builtin)
#  if __has_builtin(__builtin_amdgcn_s_wait_tensorcnt)
#    define WAIT_TENSORCNT0() __builtin_amdgcn_s_wait_tensorcnt(0)
#  else
#    define WAIT_TENSORCNT0() asm volatile("s_wait_tensorcnt 0" ::: "memory")
#  endif
#else
#  define WAIT_TENSORCNT0() asm volatile("s_wait_tensorcnt 0" ::: "memory")
#endif

// ---------------------------------------------------------------------------
// fp32 -> bf16 elementwise convert (grid-stride)
// ---------------------------------------------------------------------------
__global__ __launch_bounds__(256) void f32_to_bf16_kernel(const float* __restrict__ in,
                                                          __bf16* __restrict__ out,
                                                          long long n) {
    long long i = (long long)blockIdx.x * blockDim.x + threadIdx.x;
    long long stride = (long long)gridDim.x * blockDim.x;
    for (; i < n; i += stride) out[i] = (__bf16)in[i];
}

#if HAVE_TDM
// ---------------------------------------------------------------------------
// Issue one Tensor-Data-Mover 2D tile load: rows x cols (bf16, 2 bytes/elem)
// from a row-major tensor (row stride = stride0 elems) into contiguous LDS.
// Descriptor layout per CDNA5 ISA (D# groups 0/1).
// ---------------------------------------------------------------------------
__device__ __forceinline__ void tdm_load_tile_bf16(unsigned lds_off,
                                                   const void* gptr,
                                                   unsigned tensor_d0,
                                                   unsigned tensor_d1,
                                                   unsigned tile_d0,
                                                   unsigned tile_d1,
                                                   unsigned stride0) {
    unsigned long long ga = (unsigned long long)gptr;
    u32x4 g0;
    g0[0] = 1u;                                            // count=1, user desc
    g0[1] = lds_off;                                       // lds_addr
    g0[2] = (unsigned)(ga & 0xFFFFFFFFu);                  // global_addr[31:0]
    g0[3] = (unsigned)((ga >> 32) & 0x1FFFFFFu)            // global_addr[56:32]
          | (2u << 30);                                    // type = 2 ("image")
    i32x8 g1;
    g1[0] = (int)(1u << 16);                               // data_size=1 -> 2 bytes
    g1[1] = (int)((tensor_d0 & 0xFFFFu) << 16);            // tensor_dim0[15:0]
    g1[2] = (int)(((tensor_d0 >> 16) & 0xFFFFu)            // tensor_dim0[31:16]
          | ((tensor_d1 & 0xFFFFu) << 16));                // tensor_dim1[15:0]
    g1[3] = (int)(((tensor_d1 >> 16) & 0xFFFFu)            // tensor_dim1[31:16]
          | ((tile_d0 & 0xFFFFu) << 16));                  // tile_dim0
    g1[4] = (int)(tile_d1 & 0xFFFFu);                      // tile_dim1 (tile_dim2=0)
    g1[5] = (int)stride0;                                  // tensor_dim0_stride[31:0]
    g1[6] = 0;                                             // stride0 hi / stride1 lo
    g1[7] = 0;                                             // stride1 hi
    i32x4 z4 = {0, 0, 0, 0};
#if __clang_major__ >= 23
    i32x8 z8 = {0, 0, 0, 0, 0, 0, 0, 0};
    __builtin_amdgcn_tensor_load_to_lds(g0, g1, z4, z4, z8, 0);
#else
    __builtin_amdgcn_tensor_load_to_lds(g0, g1, z4, z4, 0);
#endif
}
#endif // HAVE_TDM

// ---------------------------------------------------------------------------
// Tiled bf16 GEMM:  out[m, n] = sum_k A[m,k] * Wt[n,k] + bias[n]
//   A  : M x K   (row major, bf16)
//   Wt : Nc x K  (row major, bf16)   (i.e. computes A @ Wt^T)
// mode 0: write bf16 to (B,H,N,D) layout  (for Q/K/V projections)
// mode 1: write fp32 to (M,Nc) row major  (final output projection)
// Block = 256 threads = 8 waves; block tile 128x64; wave tile 32x32
// (2x2 grid of 16x16x32 WMMAs). A/B tiles are DMA'd to LDS by the TDM
// (tensor_load_to_lds, TENSORcnt) with a manual-staging fallback.
// ---------------------------------------------------------------------------
__global__ __launch_bounds__(256) void gemm_bf16_wmma_kernel(
    const __bf16* __restrict__ A,
    const __bf16* __restrict__ Wt,
    const float*  __restrict__ bias,
    __bf16* __restrict__ outb,
    float*  __restrict__ outf,
    int M, int Nc, int K, int mode) {

    __shared__ __bf16 lA[TILE_M][KSTEP] __attribute__((aligned(32)));   // 8 KB
    __shared__ __bf16 lB[TILE_N][KSTEP] __attribute__((aligned(32)));   // 4 KB

    const int tid  = threadIdx.x;
    const int lane = tid & 31;
    const int wave = tid >> 5;       // 0..7
    const int wrow = wave >> 1;      // 0..3 : 32-row slice of block tile
    const int wcol = wave & 1;       // 0..1 : 32-col slice of block tile

    const int m0 = blockIdx.x * TILE_M;
    const int n0 = blockIdx.y * TILE_N;

    v8f acc[2][2] = {};

#if !HAVE_TDM
    // manual staging coordinates (fallback only)
    const int aBase = tid * 16;
    const int aRow  = aBase >> 5;
    const int aCol  = aBase & 31;
    const int bBase = tid * 8;
    const int bRow  = bBase >> 5;
    const int bCol  = bBase & 31;
#endif

    for (int k0 = 0; k0 < K; k0 += KSTEP) {
#if HAVE_TDM
        // ---- TDM DMA of the A (128x32) and B (64x32) bf16 tiles into LDS ----
        if (wave == 0) {
            tdm_load_tile_bf16((unsigned)(unsigned long long)&lA[0][0],
                               A + (size_t)m0 * K + k0,
                               (unsigned)K, (unsigned)M, KSTEP, TILE_M, (unsigned)K);
            tdm_load_tile_bf16((unsigned)(unsigned long long)&lB[0][0],
                               Wt + (size_t)n0 * K + k0,
                               (unsigned)K, (unsigned)Nc, KSTEP, TILE_N, (unsigned)K);
        }
        WAIT_TENSORCNT0();
        __syncthreads();
#else
        {
            const __bf16* src = A + (size_t)(m0 + aRow) * K + k0 + aCol;
            *(uint4*)(&lA[aRow][aCol])     = *(const uint4*)(src);
            *(uint4*)(&lA[aRow][aCol + 8]) = *(const uint4*)(src + 8);
        }
        {
            const __bf16* src = Wt + (size_t)(n0 + bRow) * K + k0 + bCol;
            *(uint4*)(&lB[bRow][bCol]) = *(const uint4*)(src);
        }
        if (k0 + KSTEP < K) {
            __builtin_prefetch(A  + (size_t)(m0 + aRow) * K + k0 + KSTEP + aCol, 0, 1);
            __builtin_prefetch(Wt + (size_t)(n0 + bRow) * K + k0 + KSTEP + bCol, 0, 1);
        }
        __syncthreads();
#endif

        // ---- load fragments from LDS ----
        v16bf afrag[2], bfrag[2];
        const int frow = lane & 15;
        const int fcol = (lane >> 4) * 16;   // each half-wave takes 16 of the 32 K values
#pragma unroll
        for (int mi = 0; mi < 2; ++mi)
            afrag[mi] = *(const v16bf*)(&lA[wrow * 32 + mi * 16 + frow][fcol]);
#pragma unroll
        for (int ni = 0; ni < 2; ++ni)
            bfrag[ni] = *(const v16bf*)(&lB[wcol * 32 + ni * 16 + frow][fcol]);

        // ---- 4 WMMAs: 2x2 grid of 16x16 tiles, K depth 32 ----
#pragma unroll
        for (int mi = 0; mi < 2; ++mi)
#pragma unroll
            for (int ni = 0; ni < 2; ++ni)
                acc[mi][ni] = __builtin_amdgcn_wmma_f32_16x16x32_bf16(
                    false, afrag[mi], false, bfrag[ni],
                    (short)0, acc[mi][ni], false, false);
        __syncthreads();
    }

    // ---- epilogue: C/D layout — VGPR r of lane l holds
    //      row_in_tile = r + 8*(l>=16), col_in_tile = l%16 ----
    const int rOff = (lane >> 4) * 8;
    const int cIn  = lane & 15;
#pragma unroll
    for (int mi = 0; mi < 2; ++mi) {
#pragma unroll
        for (int ni = 0; ni < 2; ++ni) {
            const int colBase = n0 + wcol * 32 + ni * 16 + cIn;
            const float bval = bias[colBase];
#pragma unroll
            for (int r = 0; r < 8; ++r) {
                const int row = m0 + wrow * 32 + mi * 16 + r + rOff;
                const float val = acc[mi][ni][r] + bval;
                if (mode == 0) {
                    // scatter into (B,H,N,D) bf16
                    const int b = row >> 12;          // / NSEQ
                    const int n = row & (NSEQ - 1);
                    const int h = colBase >> 6;       // / DHEAD
                    const int d = colBase & (DHEAD - 1);
                    outb[(((size_t)b * HEADS + h) * NSEQ + n) * DHEAD + d] = (__bf16)val;
                } else {
                    outf[(size_t)row * Nc + colBase] = val;
                }
            }
        }
    }
}

// ---------------------------------------------------------------------------
// Local window attention: one wave (32 lanes) per query (b,h,n).
// Each lane owns D elements {lane, lane+32}. 21-tap scores via wave32
// butterfly reduction, softmax in registers, weighted V sum.
// Writes attn-out bf16 in (B,N,H,D) so the final GEMM sees (B*N, C) rows.
// Also emits attn_last = probs[:, :, N-1, 0..10] (B*H*11 fp32).
// ---------------------------------------------------------------------------
__global__ __launch_bounds__(256) void local_attn_kernel(
    const __bf16* __restrict__ Q,   // (B,H,N,D)
    const __bf16* __restrict__ Kt,  // (B,H,N,D)
    const __bf16* __restrict__ V,   // (B,H,N,D)
    __bf16* __restrict__ O,         // (B,N,H,D)
    float* __restrict__ attn_last)  // (B*H*11)
{
    const int gq   = blockIdx.x * 8 + (threadIdx.x >> 5);   // over B*H*N
    const int lane = threadIdx.x & 31;
    if (gq >= BATCH * HEADS * NSEQ) return;

    const int n  = gq & (NSEQ - 1);
    const int bh = gq >> 12;                 // b*H + h
    const size_t base = (size_t)bh * NSEQ * DHEAD;

    const float q0 = (float)Q[base + (size_t)n * DHEAD + lane];
    const float q1 = (float)Q[base + (size_t)n * DHEAD + lane + 32];

    float sc[WINSZ];
#pragma unroll
    for (int w = 0; w < WINSZ; ++w) {
        const int pos = n - W2 + w;
        if (pos >= 0 && pos < NSEQ) {
            const __bf16* kp = Kt + base + (size_t)pos * DHEAD;
            float p = q0 * (float)kp[lane] + q1 * (float)kp[lane + 32];
#pragma unroll
            for (int off = 16; off > 0; off >>= 1)
                p += __shfl_xor(p, off, 32);
            sc[w] = p * 0.125f;              // 1/sqrt(64)
        } else {
            sc[w] = -__builtin_inff();
        }
    }

    float m = sc[0];
#pragma unroll
    for (int w = 1; w < WINSZ; ++w) m = fmaxf(m, sc[w]);
    float denom = 0.0f;
#pragma unroll
    for (int w = 0; w < WINSZ; ++w) { sc[w] = __expf(sc[w] - m); denom += sc[w]; }
    const float inv = 1.0f / denom;

    float o0 = 0.0f, o1 = 0.0f;
#pragma unroll
    for (int w = 0; w < WINSZ; ++w) {
        const int pos = n - W2 + w;
        if (pos >= 0 && pos < NSEQ) {
            const float p = sc[w] * inv;
            const __bf16* vp = V + base + (size_t)pos * DHEAD;
            o0 += p * (float)vp[lane];
            o1 += p * (float)vp[lane + 32];
        }
    }

    const int b = bh >> 4;
    const int h = bh & 15;
    const size_t ob = (((size_t)b * NSEQ + n) * HEADS + h) * DHEAD;
    O[ob + lane]      = (__bf16)o0;
    O[ob + lane + 32] = (__bf16)o1;

    if (n == NSEQ - 1 && lane == 0) {
#pragma unroll
        for (int w = 0; w <= W2; ++w)
            attn_last[bh * (W2 + 1) + w] = sc[w] * inv;
    }
}

// ---------------------------------------------------------------------------
// Host launcher
// inputs: q, Wq_w, Wq_b, Wk_w, Wk_b, Wv_w, Wv_b, Wo_w, Wo_b  (all fp32)
// d_out: [ out (B*N*C fp32) | attn_last (B*H*11 fp32) ]
// ---------------------------------------------------------------------------
extern "C" void kernel_launch(void* const* d_in, const int* in_sizes, int n_in,
                              void* d_out, int out_size, void* d_ws, size_t ws_size,
                              hipStream_t stream) {
    const float* q    = (const float*)d_in[0];
    const float* Wq_w = (const float*)d_in[1];
    const float* Wq_b = (const float*)d_in[2];
    const float* Wk_w = (const float*)d_in[3];
    const float* Wk_b = (const float*)d_in[4];
    const float* Wv_w = (const float*)d_in[5];
    const float* Wv_b = (const float*)d_in[6];
    const float* Wo_w = (const float*)d_in[7];
    const float* Wo_b = (const float*)d_in[8];

    const long long qElems = (long long)BATCH * NSEQ * CDIM;  // 8388608
    const long long wElems = (long long)CDIM * CDIM;          // 1048576

    char* ws = (char*)d_ws;
    __bf16* qb  = (__bf16*)ws;  ws += qElems * 2;
    __bf16* wqb = (__bf16*)ws;  ws += wElems * 2;
    __bf16* wkb = (__bf16*)ws;  ws += wElems * 2;
    __bf16* wvb = (__bf16*)ws;  ws += wElems * 2;
    __bf16* wob = (__bf16*)ws;  ws += wElems * 2;
    __bf16* Qb  = (__bf16*)ws;  ws += qElems * 2;
    __bf16* Kb  = (__bf16*)ws;  ws += qElems * 2;
    __bf16* Vb  = (__bf16*)ws;  ws += qElems * 2;
    __bf16* Ob  = (__bf16*)ws;  ws += qElems * 2;

    // 1) demote to bf16
    f32_to_bf16_kernel<<<1024, 256, 0, stream>>>(q, qb, qElems);
    f32_to_bf16_kernel<<<256, 256, 0, stream>>>(Wq_w, wqb, wElems);
    f32_to_bf16_kernel<<<256, 256, 0, stream>>>(Wk_w, wkb, wElems);
    f32_to_bf16_kernel<<<256, 256, 0, stream>>>(Wv_w, wvb, wElems);
    f32_to_bf16_kernel<<<256, 256, 0, stream>>>(Wo_w, wob, wElems);

    // 2) Q/K/V projections (WMMA + TDM)
    const int M = BATCH * NSEQ;                 // 8192
    dim3 gg(M / TILE_M, CDIM / TILE_N);         // (64, 16)
    gemm_bf16_wmma_kernel<<<gg, 256, 0, stream>>>(qb, wqb, Wq_b, Qb, nullptr, M, CDIM, CDIM, 0);
    gemm_bf16_wmma_kernel<<<gg, 256, 0, stream>>>(qb, wkb, Wk_b, Kb, nullptr, M, CDIM, CDIM, 0);
    gemm_bf16_wmma_kernel<<<gg, 256, 0, stream>>>(qb, wvb, Wv_b, Vb, nullptr, M, CDIM, CDIM, 0);

    // 3) local window attention (one wave per query)
    float* out_main = (float*)d_out;
    float* out_attn = out_main + qElems;
    local_attn_kernel<<<(BATCH * HEADS * NSEQ) / 8, 256, 0, stream>>>(Qb, Kb, Vb, Ob, out_attn);

    // 4) output projection (WMMA + TDM) -> fp32 d_out
    gemm_bf16_wmma_kernel<<<gg, 256, 0, stream>>>(Ob, wob, Wo_b, nullptr, out_main, M, CDIM, CDIM, 1);
}
